// RNNJANSEN_38414187495721
// MI455X (gfx1250) — matrix operand order repaired and checked by reading, not compiled, not run on hardware
//
#include <hip/hip_runtime.h>
#include <hip/hip_bf16.h>

typedef __bf16  v16bf __attribute__((ext_vector_type(16)));
typedef float   v8f   __attribute__((ext_vector_type(8)));

#define NN   200
#define HH   20
#define BB   400
#define OUTN 64
#define BUFC 64          // circular history depth (delays <= 49)
#define KPAD 224         // 7 * 32 bf16 K-chunks
#define MAXZ 16

// model constants
#define F_DT    0.0001f
#define F_UUB   500.0f
#define F_G     1000.01f
#define F_C1    135.01f
#define F_C2    108.01f
#define F_C3    33.76f
#define F_C4    33.76f
#define F_STD   250.0f
#define F_K     5.5f
#define F_A     3.25f
#define F_a     101.0f
#define F_B     22.0f
#define F_b     51.0f
#define F_CY0   5.0f
#define F_Y0    2.0f

__device__ __forceinline__ float ftanh(float x) {
    x = fminf(fmaxf(x, -15.f), 15.f);
    float e = __expf(2.f * x);
    return (e - 1.f) / (e + 1.f);
}
__device__ __forceinline__ float sigf(float x) {           // VMAX/(1+exp(R*(V0-x)))
    return 5.0f / (1.f + __expf(0.56f * (6.0f - x)));
}
__device__ __forceinline__ float satf(float x) { return 1000.f * ftanh(x * 0.001f); }

__device__ __forceinline__ unsigned short f2bf(float f) {  // RNE float->bf16 bits
    unsigned int u = __float_as_uint(f);
    u += 0x7FFFu + ((u >> 16) & 1u);
    return (unsigned short)(u >> 16);
}

// ---------- setup: wl = log1p(0.5*(exp(w_bb)*sc + sym)), accumulate ||wl||^2 ----------
__global__ __launch_bounds__(256) void k_wl(const float* __restrict__ w_bb,
                                            const float* __restrict__ sc,
                                            float* __restrict__ wn, float* __restrict__ accum) {
    __shared__ float red[256];
    int idx = blockIdx.x * 256 + threadIdx.x;
    float v = 0.f;
    if (idx < NN * NN) {
        int i = idx / NN, j = idx - i * NN;
        float wij = __expf(w_bb[i * NN + j]) * sc[i * NN + j];
        float wji = __expf(w_bb[j * NN + i]) * sc[j * NN + i];
        float wl  = log1pf(0.5f * (wij + wji));
        wn[idx] = wl;
        v = wl * wl;
    }
    red[threadIdx.x] = v;
    __syncthreads();
    for (int s = 128; s > 0; s >>= 1) {
        if (threadIdx.x < s) red[threadIdx.x] += red[threadIdx.x + s];
        __syncthreads();
    }
    if (threadIdx.x == 0) atomicAdd(accum, red[0]);
}

// ---------- setup: w_n = wl/||wl||, rowsum per row ----------
__global__ __launch_bounds__(64) void k_norm(float* __restrict__ wn,
                                             const float* __restrict__ accum,
                                             float* __restrict__ rowsum) {
    __shared__ float red[64];
    int i = blockIdx.x;
    float rinv = 1.0f / sqrtf(accum[0]);
    float s = 0.f;
    for (int j = threadIdx.x; j < NN; j += 64) {
        float v = wn[i * NN + j] * rinv;
        wn[i * NN + j] = v;
        s += v;
    }
    red[threadIdx.x] = s;
    __syncthreads();
    for (int t = 32; t > 0; t >>= 1) {
        if (threadIdx.x < t) red[threadIdx.x] += red[threadIdx.x + t];
        __syncthreads();
    }
    if (threadIdx.x == 0) rowsum[i] = red[0];
}

// ---------- setup: transposed byte delays + zero-delay sparse lists ----------
__global__ __launch_bounds__(256) void k_delays(const int* __restrict__ dist,
                                                const float* __restrict__ wn,
                                                unsigned char* __restrict__ dT,
                                                unsigned char* __restrict__ zidx,
                                                float* __restrict__ zw) {
    int i = threadIdx.x;
    if (i >= NN) return;
    int cnt = 0;
    for (int j = 0; j < NN; ++j) {
        int d = dist[j * NN + i] >> 1;          // int(dist/2.0)
        dT[i * NN + j] = (unsigned char)d;
        if (d == 0 && cnt < MAXZ) {
            zidx[i * MAXZ + cnt] = (unsigned char)j;
            zw[i * MAXZ + cnt]   = wn[i * NN + j];
            ++cnt;
        }
    }
    for (int t = cnt; t < MAXZ; ++t) { zidx[i * MAXZ + t] = 0; zw[i * MAXZ + t] = 0.f; }
}

// ---------- setup: normalize lm, center columns, pack bf16 A-fragments ----------
// A-fragment (16x32 bf16, wave32): lane L (row = L&15) half h holds
//   k = 16*(L>=16) + h   within each 32-wide K chunk.
__global__ __launch_bounds__(256) void k_lmpack(const float* __restrict__ lm,
                                                unsigned short* __restrict__ lmfrag) {
    __shared__ float rowabs[OUTN];
    __shared__ float colsum[KPAD];
    int tid = threadIdx.x;
    if (tid < OUTN) {
        float s = 0.f;
        for (int j = 0; j < NN; ++j) s += fabsf(lm[tid * NN + j]);
        rowabs[tid] = s;
    }
    __syncthreads();
    if (tid < KPAD) {
        float c = 0.f;
        if (tid < NN)
            for (int r = 0; r < OUTN; ++r) c += lm[r * NN + tid] / rowabs[r];
        colsum[tid] = c * (1.0f / OUTN);
    }
    __syncthreads();
    for (int e = tid; e < 4 * 7 * 32 * 16; e += 256) {
        int h  = e & 15;
        int L  = (e >> 4) & 31;
        int kb = (e >> 9) % 7;
        int w  = e / (16 * 32 * 7);
        int row = w * 16 + (L & 15);
        int k   = kb * 32 + ((L >= 16) ? 16 : 0) + h;
        float v = 0.f;
        if (k < NN) v = lm[row * NN + k] / rowabs[row] - colsum[k];
        lmfrag[((w * 7 + kb) * 32 + L) * 16 + h] = f2bf(v);
    }
}

// ---------- persistent recurrence kernel: 1 workgroup, 8 wave32 ----------
__global__ __launch_bounds__(256, 1) void k_main(
    const float* __restrict__ input, const float* __restrict__ noise_in,
    const float* __restrict__ hx, const float* __restrict__ hE,
    const float* __restrict__ wn, const float* __restrict__ rowsum,
    const unsigned char* __restrict__ dT, const unsigned char* __restrict__ zidx,
    const float* __restrict__ zw,
    const unsigned short* __restrict__ lmfrag, float* __restrict__ out) {

    __shared__ float  hist[NN * BUFC];                        // circular delay line (50 KB)
    __shared__ float  Mcur[2][256];                           // ping-pong live M
    __shared__ __attribute__((aligned(32))) __bf16 xlds[KPAD];// E-I, bf16, zero padded

    const int tid = threadIdx.x, n = tid;
    const int wave = tid >> 5, lane = tid & 31;

    for (int idx = tid; idx < NN * BUFC; idx += 256) {
        int node = idx >> 6, c = idx & 63;
        hist[idx] = hE[node * 500 + c];
    }

    float M = 0.f, E = 0.f, I = 0.f, Mv = 0.f, Ev = 0.f, Iv = 0.f, rs = 0.f;
    int   zj[MAXZ] = {};
    float zv[MAXZ] = {};
    if (n < NN) {
        M = hx[n * 6 + 0]; E = hx[n * 6 + 1]; I = hx[n * 6 + 2];
        Mv = hx[n * 6 + 3]; Ev = hx[n * 6 + 4]; Iv = hx[n * 6 + 5];
        rs = rowsum[n];
#pragma unroll
        for (int t = 0; t < MAXZ; ++t) { zj[t] = zidx[n * MAXZ + t]; zv[t] = zw[n * MAXZ + t]; }
        Mcur[0][n] = hE[n * 500];          // hEb[:,0] before first write is hE col 0
    }
    __syncthreads();

    int head = 0, par = 0;
    for (int b = 0; b < BB; ++b) {
        // static part of LEd for this outer step (delay > 0 columns are frozen)
        float S = 0.f;
        if (n < NN) {
            const float4* wrow4 = (const float4*)(wn + n * NN);
            const uchar4* drow4 = (const uchar4*)(dT + n * NN);
#pragma unroll 2
            for (int j4 = 0; j4 < NN / 4; ++j4) {
                uchar4 d4 = drow4[j4];
                float4 w4 = wrow4[j4];
                int jb = j4 * 4 * BUFC;
                float h0 = hist[jb + 0 * BUFC + ((head + d4.x) & (BUFC - 1))];
                float h1 = hist[jb + 1 * BUFC + ((head + d4.y) & (BUFC - 1))];
                float h2 = hist[jb + 2 * BUFC + ((head + d4.z) & (BUFC - 1))];
                float h3 = hist[jb + 3 * BUFC + ((head + d4.w) & (BUFC - 1))];
                S += (d4.x ? w4.x * h0 : 0.f) + (d4.y ? w4.y * h1 : 0.f)
                   + (d4.z ? w4.z * h2 : 0.f) + (d4.w ? w4.w * h3 : 0.f);
            }
        }

        for (int h = 0; h < HH; ++h) {
            if (n < NN) {
                float LEd = S;
#pragma unroll
                for (int t = 0; t < MAXZ; ++t)          // padding has zv==0, zj==0
                    LEd += zv[t] * Mcur[par][zj[t]];

                float u  = input[n * (HH * BB) + h * BB + b];
                float nE = noise_in[((n * HH + h) * BB + b) * 3];

                float rM = sigf(E - I);
                float rE = F_STD * nE + F_G * (LEd - rs * E) + F_C2 * sigf(F_C1 * M);
                float rI = F_C4 * sigf(F_C3 * M);

                float uM = F_UUB * ftanh(rM * (1.f / F_UUB));
                float uE = F_K * u + F_UUB * ftanh(rE * (1.f / F_UUB));
                float uI = F_UUB * ftanh(rI * (1.f / F_UUB));

                float ddM  = M  + F_DT * Mv;
                float ddE  = E  + F_DT * Ev;
                float ddI  = I  + F_DT * Iv;
                float ddMv = Mv + F_DT * (F_A * F_a * uM - 2.f * F_a * Mv - F_a * F_a * M);
                float ddEv = Ev + F_DT * (F_A * F_a * uE - 2.f * F_a * Ev - F_a * F_a * E);
                float ddIv = Iv + F_DT * (F_B * F_b * uI - 2.f * F_b * Iv - F_b * F_b * I);

                M = satf(ddM);  E = satf(ddE);  I = satf(ddI);
                Mv = satf(ddMv); Ev = satf(ddEv); Iv = satf(ddIv);

                Mcur[par ^ 1][n] = M;
            }
            __syncthreads();
            par ^= 1;
        }

        // hEb[:,0]=M, then shift: new column 0 = M
        if (n < NN) hist[n * BUFC + head] = M;
        head = (head + (BUFC - 1)) & (BUFC - 1);
        if (n < NN) hist[n * BUFC + head] = M;
        if (tid < KPAD) xlds[tid] = (tid < NN) ? (__bf16)(E - I) : (__bf16)0.f;
        __syncthreads();

        // eeg[:, b] = CY0 * (lm_t @ (E-I)) - Y0 via v_wmma_f32_16x16x32_bf16.
        // B carries x in *every* column (broadcast), so all 16 columns of D hold
        // the same matvec result -> no column masking needed at all.
        if (wave < 4) {
            v8f acc = {};
            const int khalf = (lane >> 4) << 4;   // B 32x16: lanes>=16 hold K hi-half
#pragma unroll
            for (int kb = 0; kb < 7; ++kb) {
                v16bf a  = *(const v16bf*)(lmfrag + ((wave * 7 + kb) * 32 + lane) * 16);
                v16bf xv = *(const v16bf*)(&xlds[kb * 32 + khalf]);   // vector LDS broadcast
                acc = __builtin_amdgcn_wmma_f32_16x16x32_bf16(
                    false, a, false, xv, (short)0, acc, false, false);
            }
            if ((lane & 15) == 0) {          // D col 0: lane0 rows 0..7, lane16 rows 8..15
                int rbase = wave * 16 + ((lane >> 4) << 3);
#pragma unroll
                for (int r = 0; r < 8; ++r)
                    out[(rbase + r) * BB + b] = F_CY0 * acc[r] - F_Y0;
            }
        }
        __syncthreads();

        if (n < NN && b + 1 < BB)             // gfx1250 global_prefetch_b8
            __builtin_prefetch(&input[n * (HH * BB) + (b + 1)], 0, 1);
    }

    if (n < NN) {
        float* st = out + OUTN * BB + n * 6;
        st[0] = M; st[1] = E; st[2] = I; st[3] = Mv; st[4] = Ev; st[5] = Iv;
    }
}

extern "C" void kernel_launch(void* const* d_in, const int* in_sizes, int n_in,
                              void* d_out, int out_size, void* d_ws, size_t ws_size,
                              hipStream_t stream) {
    const float* input    = (const float*)d_in[0];
    const float* noise_in = (const float*)d_in[1];
    const float* hx       = (const float*)d_in[3];
    const float* hE       = (const float*)d_in[4];
    const float* w_bb     = (const float*)d_in[5];
    const float* lm       = (const float*)d_in[6];
    const float* sc       = (const float*)d_in[7];
    const int*   dist     = (const int*)d_in[8];
    float* out = (float*)d_out;
    float* ws  = (float*)d_ws;

    float*          wn     = ws + 64;                               // 40000 f
    float*          rsum   = ws + 40064;                            // 200 f
    unsigned short* lmfrag = (unsigned short*)(ws + 40320);         // 14336 bf16
    unsigned char*  dT     = (unsigned char*)(ws + 47616);          // 40000 B
    unsigned char*  zidx   = (unsigned char*)(ws + 57664);          // 3200 B
    float*          zw     = ws + 58496;                            // 3200 f

    hipMemsetAsync(ws, 0, 64 * sizeof(float), stream);
    k_wl    <<<(NN * NN + 255) / 256, 256, 0, stream>>>(w_bb, sc, wn, ws);
    k_norm  <<<NN, 64, 0, stream>>>(wn, ws, rsum);
    k_delays<<<1, 256, 0, stream>>>(dist, wn, dT, zidx, zw);
    k_lmpack<<<1, 256, 0, stream>>>(lm, lmfrag);
    k_main  <<<1, 256, 0, stream>>>(input, noise_in, hx, hE, wn, rsum,
                                    dT, zidx, zw, lmfrag, out);
}
